// PerceiverAttention_6562710028375
// MI455X (gfx1250) — compile-verified, matrix operand order checked by reading, not run on hardware
//
#include <hip/hip_runtime.h>
#include <hip/hip_bf16.h>
#include <stdint.h>

// ---------------------------------------------------------------------------
// CDNA5 (gfx1250) wave32 WMMA implementation of PerceiverAttention.
// GEMMs + attention use v_wmma_f32_16x16x32_bf16 (fp32 accumulate).
// GEMM tile staging uses GLOBAL_LOAD_ASYNC_TO_LDS_B128 (ASYNCcnt) when the
// toolchain exposes the builtin, with a synchronous fallback otherwise.
// ---------------------------------------------------------------------------

typedef __attribute__((ext_vector_type(16))) __bf16 v16bf;
typedef __attribute__((ext_vector_type(8)))  float  v8f;
typedef __attribute__((ext_vector_type(4)))  int    v4i;

#define DEV __device__ __forceinline__

static constexpr int BQ    = 4;      // batch
static constexpr int NX    = 4096;   // n (x tokens)
static constexpr int MLAT  = 512;    // m (latents)
static constexpr int DMODEL= 1024;
static constexpr int NJ    = NX + MLAT;       // 4608
static constexpr int ROWSQ = BQ * MLAT;       // 2048
static constexpr int ROWSKV= BQ * NJ;         // 18432

DEV unsigned short f2bf(float f) {
  unsigned int u = __float_as_uint(f);
  u += 0x7FFFu + ((u >> 16) & 1u);          // round-to-nearest-even
  return (unsigned short)(u >> 16);
}
DEV unsigned int pack2(float a, float b) {
  return (unsigned int)f2bf(a) | ((unsigned int)f2bf(b) << 16);
}

union FragU { v16bf v; unsigned int u[8]; };
union U4    { uint4 q; unsigned int u[4]; unsigned short s[8]; };

// Load a 16x32 bf16 operand fragment from LDS laid out "rows x k" with pairs
// (k,k+1) contiguous. Matches the documented CDNA5 16-bit A layout:
//   lanes 0-15: rows, K in {0..7,16..23}; lanes 16-31: K in {8..15,24..31}.
// B operands use the same loader with "rows"=N columns (N-major LDS tile).
DEV v16bf load_frag(const unsigned short* base, int row0, int stride, int k0) {
  int lane = (int)(threadIdx.x & 31);
  int r  = row0 + (lane & 15);
  int kh = (lane >> 4) << 3;
  const unsigned short* p = base + r * stride + k0 + kh;
  FragU f;
#pragma unroll
  for (int i = 0; i < 4; ++i) {
    f.u[i]     = *(const unsigned int*)(p + 2 * i);
    f.u[i + 4] = *(const unsigned int*)(p + 16 + 2 * i);
  }
  return f.v;
}

DEV v8f wmma_bf16(v16bf a, v16bf b, v8f c) {
  return __builtin_amdgcn_wmma_f32_16x16x32_bf16(
      /*neg_a=*/false, a, /*neg_b=*/false, b,
      /*c_mod=*/(short)0, c, /*reuse_a=*/false, /*reuse_b=*/false);
}

DEV float half_rmax(float x) {
#pragma unroll
  for (int off = 1; off < 16; off <<= 1) x = fmaxf(x, __shfl_xor(x, off, 32));
  return x;
}
DEV float half_rsum(float x) {
#pragma unroll
  for (int off = 1; off < 16; off <<= 1) x += __shfl_xor(x, off, 32);
  return x;
}

// ---------------------------------------------------------------------------
// Async global->LDS copy (16B per lane), with sync fallback.
// Builtin signature (from hipcc diagnostic): (int4 AS1*, int4 AS3*, imm, imm)
// ---------------------------------------------------------------------------
#if __has_builtin(__builtin_amdgcn_global_load_async_to_lds_b128)
#define HAVE_ASYNC_LDS 1
DEV void cp_b128(unsigned short* l, const unsigned short* g) {
  __builtin_amdgcn_global_load_async_to_lds_b128(
      (__attribute__((address_space(1))) v4i*)(g),
      (__attribute__((address_space(3))) v4i*)(l), 0, 0);
}
#else
DEV void cp_b128(unsigned short* l, const unsigned short* g) {
  *(uint4*)l = *(const uint4*)g;
}
#endif

DEV void wait_async0() {
#if defined(HAVE_ASYNC_LDS)
#if __has_builtin(__builtin_amdgcn_s_wait_asynccnt)
  __builtin_amdgcn_s_wait_asynccnt(0);
#else
  asm volatile("s_wait_asynccnt 0x0" ::: "memory");
#endif
#endif
}

// ---------------------------------------------------------------------------
// Transpose + cast: W (K x N, f32) -> Wt (N x K, bf16) so GEMM B tiles are
// straight row segments with (k,k+1) pairs contiguous.
// ---------------------------------------------------------------------------
__global__ __launch_bounds__(256) void transpose_cast_kernel(
    const float* __restrict__ W, unsigned short* __restrict__ Wt, int K, int N) {
  __shared__ float tile[32][33];
  int k0 = (int)blockIdx.x * 32, n0 = (int)blockIdx.y * 32;
  int tx = (int)threadIdx.x & 31, ty = (int)threadIdx.x >> 5;  // ty 0..7
#pragma unroll
  for (int i = 0; i < 4; ++i)
    tile[ty * 4 + i][tx] = W[(long long)(k0 + ty * 4 + i) * N + n0 + tx];
  __syncthreads();
#pragma unroll
  for (int i = 0; i < 4; ++i) {
    int n = n0 + ty * 4 + i;
    Wt[(long long)n * K + k0 + tx] = f2bf(tile[tx][ty * 4 + i]);
  }
}

// ---------------------------------------------------------------------------
// LayerNorm over last dim (1024) + cast to bf16. One wave per row.
// out0: batched layout  base + b*out0_bstride + out0_off + r*1024
// out1: optional second copy (latents feeding the Q GEMM)
// ---------------------------------------------------------------------------
__global__ __launch_bounds__(256) void ln_cast_kernel(
    const float* __restrict__ x, const float* __restrict__ g,
    const float* __restrict__ bvec, unsigned short* __restrict__ out0,
    long long out0_bstride, long long out0_off,
    unsigned short* __restrict__ out1, int rows_per_batch) {
  int wave = (int)(threadIdx.x >> 5), lane = (int)(threadIdx.x & 31);
  long long row = (long long)blockIdx.x * 8 + wave;
  int bq = (int)(row / rows_per_batch);
  int r  = (int)(row % rows_per_batch);
  const float* xr = x + row * DMODEL;
  float vals[32];
  float s = 0.f, sq = 0.f;
#pragma unroll
  for (int i = 0; i < 8; ++i) {
    float4 v = *(const float4*)(xr + i * 128 + lane * 4);
    vals[i*4+0]=v.x; vals[i*4+1]=v.y; vals[i*4+2]=v.z; vals[i*4+3]=v.w;
    s  += v.x + v.y + v.z + v.w;
    sq += v.x*v.x + v.y*v.y + v.z*v.z + v.w*v.w;
  }
#pragma unroll
  for (int off = 1; off < 32; off <<= 1) {
    s  += __shfl_xor(s,  off, 32);
    sq += __shfl_xor(sq, off, 32);
  }
  float mean = s * (1.f / DMODEL);
  float var  = sq * (1.f / DMODEL) - mean * mean;
  float rstd = rsqrtf(var + 1e-5f);
  unsigned short* o0 = out0 + (long long)bq * out0_bstride + out0_off + (long long)r * DMODEL;
  unsigned short* o1 = out1 ? (out1 + row * DMODEL) : nullptr;
#pragma unroll
  for (int i = 0; i < 8; ++i) {
    int col = i * 128 + lane * 4;
    float4 gv = *(const float4*)(g + col);
    float4 bv = *(const float4*)(bvec + col);
    float y0 = (vals[i*4+0] - mean) * rstd * gv.x + bv.x;
    float y1 = (vals[i*4+1] - mean) * rstd * gv.y + bv.y;
    float y2 = (vals[i*4+2] - mean) * rstd * gv.z + bv.z;
    float y3 = (vals[i*4+3] - mean) * rstd * gv.w + bv.w;
    uint2 u; u.x = pack2(y0, y1); u.y = pack2(y2, y3);
    *(uint2*)(o0 + col) = u;
    if (o1) *(uint2*)(o1 + col) = u;
  }
}

// ---------------------------------------------------------------------------
// bf16 GEMM: C(f32, MxN) = A(bf16, MxK) * Bt(bf16, NxK)^T [+ bias]
// Block tile 128x128, K-step 32, 8 waves (2 M x 4 N), 64x32 per wave.
// Double-buffered LDS; tiles staged by async global->LDS DMA.
// ---------------------------------------------------------------------------
__global__ __launch_bounds__(256) void gemm_bf16_kernel(
    const unsigned short* __restrict__ A, const unsigned short* __restrict__ Bt,
    float* __restrict__ C, const float* __restrict__ bias,
    int M, int N, int K) {
  __shared__ unsigned short As[2][128 * 40];   // row-major (m x k), stride 40
  __shared__ unsigned short Bs[2][128 * 40];   // row-major (n x k), stride 40

  const int t = (int)threadIdx.x;
  const int wave = t >> 5, lane = t & 31, li = lane & 15, hh = lane >> 4;
  const int wave_m = wave >> 2, wave_n = wave & 3;
  const int m0 = (int)blockIdx.y * 128;
  const int n0 = (int)blockIdx.x * 128;

  // Tile staging: 2 threads per row, 16 elems (two b128) per thread.
  const int r_ld = t >> 1, c_ld = (t & 1) * 16;
  const unsigned short* ga_row = A  + (long long)(m0 + r_ld) * K + c_ld;
  const unsigned short* gb_row = Bt + (long long)(n0 + r_ld) * K + c_ld;
  const int l_off = r_ld * 40 + c_ld;

  v8f acc[4][2] = {};

  // prologue: stage k-step 0 into buffer 0
  cp_b128(&As[0][l_off], ga_row);     cp_b128(&As[0][l_off + 8], ga_row + 8);
  cp_b128(&Bs[0][l_off], gb_row);     cp_b128(&Bs[0][l_off + 8], gb_row + 8);

  for (int kk = 0; kk < K; kk += 32) {
    const int buf = (kk >> 5) & 1;
    wait_async0();
    __syncthreads();                  // tile kk visible to all waves
    if (kk + 32 < K) {                // stage next tile under this compute
      const unsigned short* ga = ga_row + kk + 32;
      const unsigned short* gb = gb_row + kk + 32;
      cp_b128(&As[buf ^ 1][l_off], ga);     cp_b128(&As[buf ^ 1][l_off + 8], ga + 8);
      cp_b128(&Bs[buf ^ 1][l_off], gb);     cp_b128(&Bs[buf ^ 1][l_off + 8], gb + 8);
    }
    v16bf af[4], bfr[2];
#pragma unroll
    for (int mt = 0; mt < 4; ++mt)
      af[mt] = load_frag(As[buf], wave_m * 64 + mt * 16, 40, 0);
#pragma unroll
    for (int nt = 0; nt < 2; ++nt)
      bfr[nt] = load_frag(Bs[buf], wave_n * 32 + nt * 16, 40, 0);
#pragma unroll
    for (int mt = 0; mt < 4; ++mt)
#pragma unroll
      for (int nt = 0; nt < 2; ++nt)
        acc[mt][nt] = wmma_bf16(af[mt], bfr[nt], acc[mt][nt]);
  }

#pragma unroll
  for (int nt = 0; nt < 2; ++nt) {
    const int col = n0 + wave_n * 32 + nt * 16 + li;
    const float bv = bias ? bias[col] : 0.f;
#pragma unroll
    for (int mt = 0; mt < 4; ++mt) {
      float* p = C + (long long)(m0 + wave_m * 64 + mt * 16 + hh * 8) * N + col;
#pragma unroll
      for (int v = 0; v < 8; ++v) { *p = acc[mt][nt][v] + bv; p += N; }
    }
  }
}

// ---------------------------------------------------------------------------
// RMSNorm over 64-wide head chunks of q (fp32) -> bf16 (* gamma * d^-0.5)
// One block per row; each 16-lane half handles one chunk.
// ---------------------------------------------------------------------------
__global__ __launch_bounds__(256) void rmsq_kernel(
    const float* __restrict__ q, const float* __restrict__ gq,
    unsigned short* __restrict__ out) {
  int wave = (int)(threadIdx.x >> 5), lane = (int)(threadIdx.x & 31);
  int li = lane & 15, half = lane >> 4;
  long long row = blockIdx.x;
  int chunk = wave * 2 + half;
  const float* src = q + row * 1024 + chunk * 64 + li * 4;
  float4 v = *(const float4*)src;
  float ss = half_rsum(v.x*v.x + v.y*v.y + v.z*v.z + v.w*v.w);
  float nrm = sqrtf(ss * (1.f / 64.f));
  float inv = 0.125f / fmaxf(nrm, 1e-8f);   // includes scale = d^-0.5
  float4 g4 = *(const float4*)(gq + li * 4);
  uint2 u;
  u.x = pack2(v.x * inv * g4.x, v.y * inv * g4.y);
  u.y = pack2(v.z * inv * g4.z, v.w * inv * g4.w);
  *(uint2*)(out + row * 1024 + chunk * 64 + li * 4) = u;
}

// ---------------------------------------------------------------------------
// kv post: cols [0,1024) = k -> per-64-chunk rmsnorm * gamma -> kn bf16
//          cols [1024,2048) = v -> cast -> v bf16
// ---------------------------------------------------------------------------
__global__ __launch_bounds__(256) void kvpost_kernel(
    const float* __restrict__ kv, const float* __restrict__ gk,
    unsigned short* __restrict__ kn, unsigned short* __restrict__ vb) {
  int wave = (int)(threadIdx.x >> 5), lane = (int)(threadIdx.x & 31);
  int li = lane & 15, half = lane >> 4;
  long long row = blockIdx.x;
  {
    int chunk = wave * 2 + half;
    const float* src = kv + row * 2048 + chunk * 64 + li * 4;
    float4 v = *(const float4*)src;
    float ss = half_rsum(v.x*v.x + v.y*v.y + v.z*v.z + v.w*v.w);
    float nrm = sqrtf(ss * (1.f / 64.f));
    float inv = 1.f / fmaxf(nrm, 1e-8f);
    float4 g4 = *(const float4*)(gk + li * 4);
    uint2 u;
    u.x = pack2(v.x * inv * g4.x, v.y * inv * g4.y);
    u.y = pack2(v.z * inv * g4.z, v.w * inv * g4.w);
    *(uint2*)(kn + row * 1024 + chunk * 64 + li * 4) = u;
  }
  {
    int col = (int)threadIdx.x * 4;
    float4 v = *(const float4*)(kv + row * 2048 + 1024 + col);
    uint2 u; u.x = pack2(v.x, v.y); u.y = pack2(v.z, v.w);
    *(uint2*)(vb + row * 1024 + col) = u;
  }
}

// ---------------------------------------------------------------------------
// Flash attention. Grid (4 i-tiles, 16 heads, 4 batches), 256 threads.
// Each block: 128 q rows; each wave owns 16 rows (matches WMMA C layout:
// lanes 0-15 -> row v, lanes 16-31 -> row v+8 for accumulator VGPR v).
// ---------------------------------------------------------------------------
__global__ __launch_bounds__(256) void flash_kernel(
    const unsigned short* __restrict__ Qn, const unsigned short* __restrict__ Kn,
    const unsigned short* __restrict__ Vb, const unsigned char* __restrict__ mask,
    unsigned short* __restrict__ Out) {
  __shared__ unsigned short Qs[128 * 72];   // 128 rows x 64 d, stride 72
  __shared__ unsigned short Ks[128 * 72];   // 128 j   x 64 d, stride 72 (B for QK^T)
  __shared__ unsigned short Vs[64 * 130];   // 64 d x 128 j (transposed; B for PV)
  __shared__ unsigned short Ps[128 * 130];  // 128 rows x 128 j bf16 probs

  const int i0 = (int)blockIdx.x * 128;
  const int h  = (int)blockIdx.y;
  const int bq = (int)blockIdx.z;
  const int t = (int)threadIdx.x;
  const int wave = t >> 5, lane = t & 31, li = lane & 15, hh = lane >> 4;

  // ---- load Q tile (128 x 64) ----
  {
    int r = t >> 1, seg = (t & 1) * 32;
    const unsigned short* src =
        Qn + (long long)(bq * MLAT + i0 + r) * 1024 + h * 64 + seg;
    uint4 q0 = *(const uint4*)(src);
    uint4 q1 = *(const uint4*)(src + 8);
    uint4 q2 = *(const uint4*)(src + 16);
    uint4 q3 = *(const uint4*)(src + 24);
    *(uint4*)(Qs + r * 72 + seg)      = q0;
    *(uint4*)(Qs + r * 72 + seg + 8)  = q1;
    *(uint4*)(Qs + r * 72 + seg + 16) = q2;
    *(uint4*)(Qs + r * 72 + seg + 24) = q3;
  }

  float mstat[8], lstat[8];
#pragma unroll
  for (int v = 0; v < 8; ++v) { mstat[v] = -3.0e38f; lstat[v] = 0.f; }
  v8f oacc[4] = {};

  for (int j0 = 0; j0 < NJ; j0 += 128) {
    __syncthreads();   // prev-iter compute + Q stores visible
    // ---- K tile: row-major (j x d) ----
    {
      int r = t >> 1, seg = (t & 1) * 32;
      const unsigned short* src =
          Kn + (long long)(bq * NJ + j0 + r) * 1024 + h * 64 + seg;
      uint4 k0 = *(const uint4*)(src);
      uint4 k1 = *(const uint4*)(src + 8);
      uint4 k2 = *(const uint4*)(src + 16);
      uint4 k3 = *(const uint4*)(src + 24);
      *(uint4*)(Ks + r * 72 + seg)      = k0;
      *(uint4*)(Ks + r * 72 + seg + 8)  = k1;
      *(uint4*)(Ks + r * 72 + seg + 16) = k2;
      *(uint4*)(Ks + r * 72 + seg + 24) = k3;
    }
    // ---- V tile transposed into (d x j), pairs along j ----
    {
      int jp = t >> 2, d0 = (t & 3) * 16;
      const unsigned short* v0 =
          Vb + (long long)(bq * NJ + j0 + 2 * jp) * 1024 + h * 64 + d0;
      const unsigned short* v1 = v0 + 1024;
      U4 a0, a1, b0, b1;
      a0.q = *(const uint4*)(v0);      a1.q = *(const uint4*)(v0 + 8);
      b0.q = *(const uint4*)(v1);      b1.q = *(const uint4*)(v1 + 8);
#pragma unroll
      for (int i = 0; i < 8; ++i) {
        *(unsigned int*)(Vs + (d0 + i) * 130 + 2 * jp) =
            (unsigned int)a0.s[i] | ((unsigned int)b0.s[i] << 16);
        *(unsigned int*)(Vs + (d0 + 8 + i) * 130 + 2 * jp) =
            (unsigned int)a1.s[i] | ((unsigned int)b1.s[i] << 16);
      }
    }
    __syncthreads();

    // ---- S = Q K^T  (wave rows 16w..16w+15, full 128 cols) ----
    v8f s[8] = {};
#pragma unroll
    for (int kc = 0; kc < 64; kc += 32) {
      v16bf aq = load_frag(Qs, wave * 16, 72, kc);
#pragma unroll
      for (int jt = 0; jt < 8; ++jt) {
        v16bf bk = load_frag(Ks, jt * 16, 72, kc);
        s[jt] = wmma_bf16(aq, bk, s[jt]);
      }
    }

    // ---- mask (padded mask: latents always valid) ----
#pragma unroll
    for (int jt = 0; jt < 8; ++jt) {
      int jg = j0 + jt * 16 + li;
      bool ok = (jg >= NX) || (mask[(long long)bq * NX + jg] != 0);
      if (!ok) {
#pragma unroll
        for (int v = 0; v < 8; ++v) s[jt][v] = -3.0e38f;
      }
    }

    // ---- online softmax (stats replicated within each 16-lane half) ----
    float sc[8];
#pragma unroll
    for (int v = 0; v < 8; ++v) {
      float mx = s[0][v];
#pragma unroll
      for (int jt = 1; jt < 8; ++jt) mx = fmaxf(mx, s[jt][v]);
      mx = half_rmax(mx);
      float mnew = fmaxf(mstat[v], mx);
      sc[v] = __expf(mstat[v] - mnew);
      mstat[v] = mnew;
    }
#pragma unroll
    for (int jt = 0; jt < 8; ++jt)
#pragma unroll
      for (int v = 0; v < 8; ++v) s[jt][v] = __expf(s[jt][v] - mstat[v]);
#pragma unroll
    for (int v = 0; v < 8; ++v) {
      float rs = 0.f;
#pragma unroll
      for (int jt = 0; jt < 8; ++jt) rs += s[jt][v];
      rs = half_rsum(rs);
      lstat[v] = lstat[v] * sc[v] + rs;
    }

    // ---- write P (bf16) to this wave's own LDS rows ----
#pragma unroll
    for (int jt = 0; jt < 8; ++jt)
#pragma unroll
      for (int v = 0; v < 8; ++v)
        Ps[(wave * 16 + v + hh * 8) * 130 + jt * 16 + li] = f2bf(s[jt][v]);

    // ---- rescale O, then O += P V (same wave reads its own P rows) ----
#pragma unroll
    for (int nt = 0; nt < 4; ++nt)
#pragma unroll
      for (int v = 0; v < 8; ++v) oacc[nt][v] *= sc[v];
#pragma unroll
    for (int kc = 0; kc < 128; kc += 32) {
      v16bf ap = load_frag(Ps, wave * 16, 130, kc);
#pragma unroll
      for (int nt = 0; nt < 4; ++nt) {
        v16bf bv = load_frag(Vs, nt * 16, 130, kc);
        oacc[nt] = wmma_bf16(ap, bv, oacc[nt]);
      }
    }
  }

  // ---- epilogue: O / l -> bf16 at (b*m + row, h*64 + d) ----
  float inv[8];
#pragma unroll
  for (int v = 0; v < 8; ++v) inv[v] = 1.f / lstat[v];
#pragma unroll
  for (int nt = 0; nt < 4; ++nt) {
#pragma unroll
    for (int v = 0; v < 8; ++v) {
      int row = bq * MLAT + i0 + wave * 16 + v + hh * 8;
      int col = h * 64 + nt * 16 + li;
      Out[(long long)row * 1024 + col] = f2bf(oacc[nt][v] * inv[v]);
    }
  }
}

// ---------------------------------------------------------------------------
// Launcher
// ---------------------------------------------------------------------------
extern "C" void kernel_launch(void* const* d_in, const int* in_sizes, int n_in,
                              void* d_out, int out_size, void* d_ws, size_t ws_size,
                              hipStream_t stream) {
  (void)in_sizes; (void)n_in; (void)out_size; (void)ws_size;
  const float* x       = (const float*)d_in[0];
  const float* latents = (const float*)d_in[1];
  const unsigned char* mask = (const unsigned char*)d_in[2];
  const float* ln_x_g  = (const float*)d_in[3];
  const float* ln_x_b  = (const float*)d_in[4];
  const float* ln_l_g  = (const float*)d_in[5];
  const float* ln_l_b  = (const float*)d_in[6];
  const float* qn_g    = (const float*)d_in[7];
  const float* kn_g    = (const float*)d_in[8];
  const float* Wq      = (const float*)d_in[9];
  const float* Wkv     = (const float*)d_in[10];
  const float* Wout    = (const float*)d_in[11];
  const float* bout    = (const float*)d_in[12];
  float* out = (float*)d_out;

  char* ws = (char*)d_ws;
  size_t off = 0;
  auto take = [&](size_t bytes) -> char* {
    char* p = ws + off;
    off = (off + bytes + 255) & ~(size_t)255;
    return p;
  };
  unsigned short* kvin   = (unsigned short*)take((size_t)ROWSKV * 1024 * 2); // [xn;ln] bf16
  unsigned short* lnq    = (unsigned short*)take((size_t)ROWSQ  * 1024 * 2); // ln bf16
  unsigned short* wq_t   = (unsigned short*)take((size_t)1024 * 1024 * 2);   // Wq^T  (N x K)
  unsigned short* wkv_t  = (unsigned short*)take((size_t)2048 * 1024 * 2);   // Wkv^T (N x K)
  unsigned short* wout_t = (unsigned short*)take((size_t)1024 * 1024 * 2);   // Wout^T(N x K)
  float*          qf32   = (float*)take((size_t)ROWSQ  * 1024 * 4);
  float*          kvf32  = (float*)take((size_t)ROWSKV * 2048 * 4);
  unsigned short* qn_bf  = (unsigned short*)take((size_t)ROWSQ  * 1024 * 2);
  unsigned short* kn_bf  = (unsigned short*)take((size_t)ROWSKV * 1024 * 2);
  unsigned short* v_bf   = (unsigned short*)take((size_t)ROWSKV * 1024 * 2);
  unsigned short* att_bf = (unsigned short*)take((size_t)ROWSQ  * 1024 * 2);

  // 1) weights -> transposed bf16 (N-major, k-pairs contiguous)
  transpose_cast_kernel<<<dim3(32, 32), 256, 0, stream>>>(Wq,   wq_t,   1024, 1024);
  transpose_cast_kernel<<<dim3(32, 64), 256, 0, stream>>>(Wkv,  wkv_t,  1024, 2048);
  transpose_cast_kernel<<<dim3(32, 32), 256, 0, stream>>>(Wout, wout_t, 1024, 1024);

  // 2) layernorms -> bf16 (x into kv-input rows [0,n); latents into [n,n+m) + lnq)
  ln_cast_kernel<<<(BQ * NX) / 8, 256, 0, stream>>>(
      x, ln_x_g, ln_x_b, kvin, (long long)NJ * 1024, 0LL, nullptr, NX);
  ln_cast_kernel<<<(BQ * MLAT) / 8, 256, 0, stream>>>(
      latents, ln_l_g, ln_l_b, kvin, (long long)NJ * 1024,
      (long long)NX * 1024, lnq, MLAT);

  // 3) q = ln @ Wq        (2048 x 1024 x 1024)
  gemm_bf16_kernel<<<dim3(1024 / 128, ROWSQ / 128), 256, 0, stream>>>(
      lnq, wq_t, qf32, nullptr, ROWSQ, 1024, 1024);

  // 4) kv = [xn;ln] @ Wkv (18432 x 2048 x 1024)  -- dominant GEMM
  gemm_bf16_kernel<<<dim3(2048 / 128, ROWSKV / 128), 256, 0, stream>>>(
      kvin, wkv_t, kvf32, nullptr, ROWSKV, 2048, 1024);

  // 5) rmsnorm(q)*scale -> bf16 ; rmsnorm(k) -> bf16, cast v -> bf16
  rmsq_kernel<<<ROWSQ, 256, 0, stream>>>(qf32, qn_g, qn_bf);
  kvpost_kernel<<<ROWSKV, 256, 0, stream>>>(kvf32, kn_g, kn_bf, v_bf);

  // 6) flash attention per (b, h, 128-row q tile)
  flash_kernel<<<dim3(MLAT / 128, 16, BQ), 256, 0, stream>>>(
      qn_bf, kn_bf, v_bf, mask, att_bf);

  // 7) out = attn_out @ Wout + bout -> fp32
  gemm_bf16_kernel<<<dim3(1024 / 128, ROWSQ / 128), 256, 0, stream>>>(
      att_bf, wout_t, out, bout, ROWSQ, 1024, 1024);
}
